// STALSTM_76321568850274
// MI455X (gfx1250) — matrix operand order repaired and checked by reading, not compiled, not run on hardware
//
#include <hip/hip_runtime.h>
#include <hip/hip_bf16.h>

typedef __attribute__((ext_vector_type(16))) __bf16 v16bf;
typedef __attribute__((ext_vector_type(8)))  __bf16 v8bf;
typedef __attribute__((ext_vector_type(8)))  float  v8f;

// ---------------------------------------------------------------------------
// helpers
// ---------------------------------------------------------------------------
__device__ inline float b2f(__hip_bfloat16 h) { return __bfloat162float(h); }
__device__ inline __hip_bfloat16 f2b(float f) { return __float2bfloat16(f); }
__device__ inline float sigm(float v) { return 1.0f / (1.0f + expf(-v)); }

__device__ inline v8f wmma_bf16(v16bf a, v16bf b, v8f c) {
    // D = A(16x32) * B(32x16) + C, fp32 accumulate
    return __builtin_amdgcn_wmma_f32_16x16x32_bf16(
        /*neg_a=*/false, a, /*neg_b=*/false, b,
        /*c_mod=*/(short)0, c, /*reuse_a=*/false, /*reuse_b=*/false);
}

// A fragment: row-major (M x ld) bf16, rows m..m+15, K k..k+31.
// lane l: row = m + (l&15); elems 0..7 = K+base..K+base+7, elems 8..15 = +16
__device__ inline v16bf load_a_frag(const __bf16* base, int ld, int m, int k, int lane) {
    int mm = m + (lane & 15);
    int kb = k + ((lane >> 4) << 3);          // +8 for lanes 16..31
    const __bf16* p = base + (size_t)mm * ld + kb;
    v8bf lo = *(const v8bf*)p;
    v8bf hi = *(const v8bf*)(p + 16);
    v16bf a;
#pragma unroll
    for (int i = 0; i < 8; ++i) { a[i] = lo[i]; a[i + 8] = hi[i]; }
    return a;
}

// B fragment: W row-major (N x ld) bf16 == B^T; cols n..n+15, K k..k+31.
// lane l: col = n + (l&15); elems 0..7 = K+base..K+base+7, elems 8..15 = +16, base = (l>=16)*8
__device__ inline v16bf load_b_frag(const __bf16* base, int ld, int n, int k, int lane) {
    int nn = n + (lane & 15);
    int kb = k + ((lane >> 4) << 3);          // +8 for lanes 16..31
    const __bf16* p = base + (size_t)nn * ld + kb;
    v8bf lo = *(const v8bf*)p;
    v8bf hi = *(const v8bf*)(p + 16);
    v16bf b;
#pragma unroll
    for (int i = 0; i < 8; ++i) { b[i] = lo[i]; b[i + 8] = hi[i]; }
    return b;
}

// ---------------------------------------------------------------------------
// cast fp32 -> bf16 with column zero-padding (cols -> dcols)
// ---------------------------------------------------------------------------
__global__ void k_cast_pad(const float* __restrict__ src, __hip_bfloat16* __restrict__ dst,
                           int rows, int cols, int dcols) {
    int idx = blockIdx.x * 256 + threadIdx.x;
    int total = rows * dcols;
    if (idx >= total) return;
    int r = idx / dcols, c = idx - r * dcols;
    float v = (c < cols) ? src[(size_t)r * cols + c] : 0.0f;
    dst[idx] = f2b(v);
}

// xw = x * alpha_b (broadcast over T), bf16 + pad 75->96
__global__ void k_scale(const float* __restrict__ x, const float* __restrict__ alpha,
                        __hip_bfloat16* __restrict__ dst, int total) {
    int idx = blockIdx.x * 256 + threadIdx.x;
    if (idx >= total) return;
    int r = idx / 96, c = idx - r * 96;
    float v = 0.0f;
    if (c < 75) {
        int b = r & 63;                       // r = t*64 + b
        v = x[(size_t)r * 75 + c] * alpha[b * 75 + c];
    }
    dst[idx] = f2b(v);
}

// ---------------------------------------------------------------------------
// Generic WMMA GEMM:  out(M x N) = A(M x K)bf16 * W(N x K)bf16^T + bias0 + bias1
// block = 1 wave (32 threads); each wave computes a 64x64 tile (4x4 WMMA tiles)
// grid = (N/64, M/64). out fp32 or bf16.
// ---------------------------------------------------------------------------
__global__ void __launch_bounds__(32)
k_gemm(const __hip_bfloat16* __restrict__ A_, int lda,
       const __hip_bfloat16* __restrict__ W_, int ldw,
       const float* __restrict__ bias0, const float* __restrict__ bias1,
       void* __restrict__ out, int ldo, int K, int outBf16) {
    const int lane = threadIdx.x;
    const int n0 = blockIdx.x * 64;
    const int m0 = blockIdx.y * 64;
    const __bf16* A = (const __bf16*)A_;
    const __bf16* W = (const __bf16*)W_;

    v8f acc[4][4];
#pragma unroll
    for (int nt = 0; nt < 4; ++nt) {
        int n = n0 + nt * 16 + (lane & 15);
        float bv = (bias0 ? bias0[n] : 0.0f) + (bias1 ? bias1[n] : 0.0f);
        v8f z;
#pragma unroll
        for (int r = 0; r < 8; ++r) z[r] = bv;
#pragma unroll
        for (int mt = 0; mt < 4; ++mt) acc[mt][nt] = z;
    }
    for (int kk = 0; kk < K; kk += 32) {
        v16bf a[4], b[4];
#pragma unroll
        for (int mt = 0; mt < 4; ++mt) a[mt] = load_a_frag(A, lda, m0 + mt * 16, kk, lane);
#pragma unroll
        for (int nt = 0; nt < 4; ++nt) b[nt] = load_b_frag(W, ldw, n0 + nt * 16, kk, lane);
#pragma unroll
        for (int mt = 0; mt < 4; ++mt)
#pragma unroll
            for (int nt = 0; nt < 4; ++nt)
                acc[mt][nt] = wmma_bf16(a[mt], b[nt], acc[mt][nt]);
    }
    const int sel = lane >> 4;
#pragma unroll
    for (int mt = 0; mt < 4; ++mt)
#pragma unroll
        for (int nt = 0; nt < 4; ++nt) {
            int n = n0 + nt * 16 + (lane & 15);
            int mb = m0 + mt * 16 + sel * 8;      // VGPR r holds row mb + r
#pragma unroll
            for (int r = 0; r < 8; ++r) {
                size_t o = (size_t)(mb + r) * ldo + n;
                if (outBf16) ((__hip_bfloat16*)out)[o] = f2b(acc[mt][nt][r]);
                else         ((float*)out)[o] = acc[mt][nt][r];
            }
        }
}

// ---------------------------------------------------------------------------
// BiLSTM recurrence: blockIdx.x = direction (0 fwd, 1 bwd), 1024 threads.
// Per step: phase1 (32 waves) g = h @ Whh^T via WMMA, h from LDS (bf16),
//           phase2 adds xproj, applies gates, c in registers, h -> LDS + y.
// Wave w owns M-row tile (w>>3) and 16 N-tiles [(w&7)*16, +16), processed in
// two groups of 8 so each A fragment (from LDS) is reused by 8 WMMAs.
// Whh: (dir, 2048, 512) bf16. xproj: (dir, T, 64, 2048) bf16 (has biases).
// gbuf: (dir, 64, 2048) fp32 scratch. y: (T, 64, 1024) bf16 or null.
// hn: (2, 64, 512) fp32 or null. h0/c0: (2, 64, 512) fp32 slices.
// ---------------------------------------------------------------------------
__global__ void __launch_bounds__(1024, 1)
k_lstm(const __hip_bfloat16* __restrict__ Whh,
       const __hip_bfloat16* __restrict__ xproj,
       const float* __restrict__ h0, const float* __restrict__ c0,
       float* __restrict__ gbuf,
       __hip_bfloat16* __restrict__ y,
       float* __restrict__ hn, int T_) {
    __shared__ __hip_bfloat16 hbuf[64 * 512];          // 64 KB
    const int dir  = blockIdx.x;
    const int tid  = threadIdx.x;
    const int lane = tid & 31;
    const int wave = tid >> 5;
    const int mt   = wave >> 3;                        // M-row tile (0..3)
    const int nb   = (wave & 7) * 16;                  // first N-tile of this wave

    const __bf16* Wd = (const __bf16*)Whh + (size_t)dir * 2048 * 512;
    float* gb = gbuf + (size_t)dir * 64 * 2048;
    const __bf16* hbf = (const __bf16*)hbuf;

    // init h (LDS, bf16) and c (registers)
    float creg[32];
    {
        const float* h0d = h0 + (size_t)dir * 64 * 512;
        const float* c0d = c0 + (size_t)dir * 64 * 512;
#pragma unroll
        for (int j = 0; j < 32; ++j) {
            int e = tid + (j << 10);
            hbuf[e] = f2b(h0d[e]);
            creg[j] = c0d[e];
        }
    }
    __syncthreads();

    for (int step = 0; step < T_; ++step) {
        const int t = dir ? (T_ - 1 - step) : step;

        // -------- phase 1: g = h @ Whh^T  (64 x 2048, K = 512) --------
        for (int g = 0; g < 2; ++g) {
            v8f zero = {};
            v8f acc[8];
#pragma unroll
            for (int j = 0; j < 8; ++j) acc[j] = zero;
            for (int kf = 0; kf < 16; ++kf) {
                v16bf a = load_a_frag(hbf, 512, mt * 16, kf * 32, lane);
#pragma unroll
                for (int j = 0; j < 8; ++j) {
                    int nt = nb + g * 8 + j;
                    v16bf b = load_b_frag(Wd, 512, nt * 16, kf * 32, lane);
                    acc[j] = wmma_bf16(a, b, acc[j]);
                }
            }
            int mrow = mt * 16 + ((lane >> 4) << 3);
#pragma unroll
            for (int j = 0; j < 8; ++j) {
                int n = (nb + g * 8 + j) * 16 + (lane & 15);
#pragma unroll
                for (int r = 0; r < 8; ++r)
                    gb[(size_t)(mrow + r) * 2048 + n] = acc[j][r];
            }
        }
        __threadfence();
        __syncthreads();
        __threadfence();

        // -------- phase 2: gates, cell update --------
        const __hip_bfloat16* xp = xproj + ((size_t)dir * T_ + t) * 64 * 2048;
#pragma unroll
        for (int j = 0; j < 32; ++j) {
            int e  = tid + (j << 10);
            int bb = e >> 9, hh = e & 511;
            size_t row = (size_t)bb * 2048;
            float gi = gb[row +        hh] + b2f(xp[row +        hh]);
            float gf = gb[row +  512 + hh] + b2f(xp[row +  512 + hh]);
            float gg = gb[row + 1024 + hh] + b2f(xp[row + 1024 + hh]);
            float go = gb[row + 1536 + hh] + b2f(xp[row + 1536 + hh]);
            float cn = sigm(gf) * creg[j] + sigm(gi) * tanhf(gg);
            float hv = sigm(go) * tanhf(cn);
            creg[j] = cn;
            hbuf[e] = f2b(hv);
            if (y)  y[((size_t)t * 64 + bb) * 1024 + (size_t)dir * 512 + hh] = f2b(hv);
            if (hn && step == T_ - 1) hn[(size_t)dir * 64 * 512 + e] = hv;
        }
        __syncthreads();
    }
}

// ---------------------------------------------------------------------------
// Spatial attention: one block per batch element, fp32 VALU (tiny).
// hn (2,64,512) -> alpha_b (64,75)
// ---------------------------------------------------------------------------
__global__ void __launch_bounds__(256)
k_attn(const float* __restrict__ hn,
       const float* __restrict__ W1, const float* __restrict__ b1,
       const float* __restrict__ W2, const float* __restrict__ b2,
       float* __restrict__ alpha_b) {
    __shared__ float t1[1024];
    __shared__ float ev[160];
    __shared__ float red[256];
    __shared__ float Ssum, Scnt, Ssel;
    const int b = blockIdx.x, tid = threadIdx.x;

    for (int idx = tid; idx < 1024; idx += 256) {
        int d = idx >> 9, hcol = idx & 511;
        const float* hrow = hn + ((size_t)d * 64 + b) * 512;
        const float* wrow = W1 + (size_t)hcol * 512;
        float s = 0.0f;
        for (int k = 0; k < 512; ++k) s += hrow[k] * wrow[k];
        t1[idx] = tanhf(s + b1[hcol]);
    }
    __syncthreads();
    for (int idx = tid; idx < 150; idx += 256) {
        int d = idx / 75, jj = idx - d * 75;
        const float* wrow = W2 + (size_t)jj * 512;
        float s = 0.0f;
        for (int k = 0; k < 512; ++k) s += t1[d * 512 + k] * wrow[k];
        ev[idx] = expf(s + b2[jj]);
    }
    __syncthreads();
    // sum(e) over (dir, J)
    red[tid] = (tid < 150) ? ev[tid] : 0.0f;
    __syncthreads();
    for (int s = 128; s > 0; s >>= 1) { if (tid < s) red[tid] += red[tid + s]; __syncthreads(); }
    if (tid == 0) Ssum = red[0];
    __syncthreads();
    // alpha, mask stats
    float cnt = 0.0f, sel = 0.0f;
    if (tid < 150) {
        float a = ev[tid] / Ssum;
        ev[tid] = a;
        if (a >= 0.1f) { cnt = 1.0f; sel = a; }
    }
    __syncthreads();
    red[tid] = cnt; __syncthreads();
    for (int s = 128; s > 0; s >>= 1) { if (tid < s) red[tid] += red[tid + s]; __syncthreads(); }
    if (tid == 0) Scnt = red[0];
    __syncthreads();
    red[tid] = sel; __syncthreads();
    for (int s = 128; s > 0; s >>= 1) { if (tid < s) red[tid] += red[tid + s]; __syncthreads(); }
    if (tid == 0) Ssel = red[0];
    __syncthreads();
    float sm = Ssel / fmaxf(Scnt, 1.0f);
    if (tid < 150) { float a = ev[tid]; if (a >= 0.1f) a = sm; ev[tid] = a; }
    __syncthreads();
    if (tid < 75) alpha_b[(size_t)b * 75 + tid] = 0.5f * (ev[tid] + ev[75 + tid]);
}

// ---------------------------------------------------------------------------
// host-side orchestration
// ---------------------------------------------------------------------------
extern "C" void kernel_launch(void* const* d_in, const int* in_sizes, int n_in,
                              void* d_out, int out_size, void* d_ws, size_t ws_size,
                              hipStream_t stream) {
    (void)in_sizes; (void)n_in; (void)out_size; (void)ws_size;
    const float* x     = (const float*)d_in[0];
    const float* h0    = (const float*)d_in[1];
    const float* c0    = (const float*)d_in[2];
    const float* saWih = (const float*)d_in[3];
    const float* saWhh = (const float*)d_in[4];
    const float* sabih = (const float*)d_in[5];
    const float* sabhh = (const float*)d_in[6];
    const float* m0Wih = (const float*)d_in[7];
    const float* m0Whh = (const float*)d_in[8];
    const float* m0bih = (const float*)d_in[9];
    const float* m0bhh = (const float*)d_in[10];
    const float* mLWih = (const float*)d_in[11];
    const float* mLWhh = (const float*)d_in[12];
    const float* mLbih = (const float*)d_in[13];
    const float* mLbhh = (const float*)d_in[14];
    const float* f1W   = (const float*)d_in[15];
    const float* f1b   = (const float*)d_in[16];
    const float* f2W   = (const float*)d_in[17];
    const float* f2b   = (const float*)d_in[18];
    const float* fcW   = (const float*)d_in[19];
    const float* fcb   = (const float*)d_in[20];

    const size_t TB = 8192;                    // T*B
    char* ws = (char*)d_ws;
    size_t off = 0;
    auto take = [&](size_t bytes) -> void* {
        off = (off + 255) & ~(size_t)255;
        void* p = ws + off;
        off += bytes;
        return p;
    };
    __hip_bfloat16* xb    = (__hip_bfloat16*)take(TB * 96 * 2);
    __hip_bfloat16* xwb   = (__hip_bfloat16*)take(TB * 96 * 2);
    __hip_bfloat16* Wsa   = (__hip_bfloat16*)take((size_t)2 * 2048 * 96 * 2);
    __hip_bfloat16* Wm0   = (__hip_bfloat16*)take((size_t)2 * 2048 * 96 * 2);
    __hip_bfloat16* WmL   = (__hip_bfloat16*)take((size_t)4 * 2048 * 1024 * 2);
    __hip_bfloat16* WhhB  = (__hip_bfloat16*)take((size_t)8 * 2048 * 512 * 2);
    __hip_bfloat16* Wfc   = (__hip_bfloat16*)take((size_t)512 * 1024 * 2);
    __hip_bfloat16* xproj = (__hip_bfloat16*)take((size_t)2 * TB * 2048 * 2);
    float*          gbuf  = (float*)take((size_t)2 * 64 * 2048 * 4);
    float*          hnb   = (float*)take((size_t)2 * 64 * 512 * 4);
    float*          alpha = (float*)take((size_t)64 * 75 * 4);
    __hip_bfloat16* y0    = (__hip_bfloat16*)take(TB * 1024 * 2);
    __hip_bfloat16* y1    = (__hip_bfloat16*)take(TB * 1024 * 2);

    auto cast = [&](const float* s, __hip_bfloat16* d, int rows, int cols, int dcols) {
        int tot = rows * dcols;
        k_cast_pad<<<(tot + 255) / 256, 256, 0, stream>>>(s, d, rows, cols, dcols);
    };
    auto gemm = [&](const __hip_bfloat16* A, int lda, const __hip_bfloat16* W, int ldw,
                    const float* bb0, const float* bb1, void* out, int ldo,
                    int M, int N, int K, int obf) {
        k_gemm<<<dim3(N / 64, M / 64), 32, 0, stream>>>(A, lda, W, ldw, bb0, bb1, out, ldo, K, obf);
    };

    // 1. casts (weights + activations)
    cast(x,     xb,  8192, 75, 96);
    cast(saWih, Wsa, 4096, 75, 96);
    cast(m0Wih, Wm0, 4096, 75, 96);
    cast(mLWih, WmL, 8192, 1024, 1024);
    cast(saWhh, WhhB,                       4096, 512, 512);
    cast(m0Whh, WhhB + (size_t)4096 * 512,  4096, 512, 512);
    cast(mLWhh, WhhB + (size_t)8192 * 512,  8192, 512, 512);
    cast(fcW,   Wfc,  512, 1024, 1024);

    // 2. spatial-attention LSTM
    for (int d = 0; d < 2; ++d)
        gemm(xb, 96, Wsa + (size_t)d * 2048 * 96, 96, sabih + d * 2048, sabhh + d * 2048,
             xproj + (size_t)d * TB * 2048, 2048, 8192, 2048, 96, 1);
    k_lstm<<<2, 1024, 0, stream>>>(WhhB, xproj, h0, c0, gbuf, nullptr, hnb, 128);
    k_attn<<<64, 256, 0, stream>>>(hnb, f1W, f1b, f2W, f2b, alpha);
    k_scale<<<(8192 * 96 + 255) / 256, 256, 0, stream>>>(x, alpha, xwb, 8192 * 96);

    // 3. main LSTM layer 0 (input = alpha-scaled x)
    for (int d = 0; d < 2; ++d)
        gemm(xwb, 96, Wm0 + (size_t)d * 2048 * 96, 96, m0bih + d * 2048, m0bhh + d * 2048,
             xproj + (size_t)d * TB * 2048, 2048, 8192, 2048, 96, 1);
    k_lstm<<<2, 1024, 0, stream>>>(WhhB + (size_t)1 * 2 * 2048 * 512, xproj, h0, c0,
                                   gbuf, y0, nullptr, 128);

    // 4. main LSTM layers 1,2 (input = previous y, K=1024)
    __hip_bfloat16* yin = y0;
    __hip_bfloat16* yout = y1;
    for (int l = 0; l < 2; ++l) {
        for (int d = 0; d < 2; ++d)
            gemm(yin, 1024, WmL + (size_t)(l * 2 + d) * 2048 * 1024, 1024,
                 mLbih + (l * 2 + d) * 2048, mLbhh + (l * 2 + d) * 2048,
                 xproj + (size_t)d * TB * 2048, 2048, 8192, 2048, 1024, 1);
        k_lstm<<<2, 1024, 0, stream>>>(WhhB + (size_t)(2 + l) * 2 * 2048 * 512, xproj,
                                       h0 + (size_t)(2 + 2 * l) * 64 * 512,
                                       c0 + (size_t)(2 + 2 * l) * 64 * 512,
                                       gbuf, yout, nullptr, 128);
        __hip_bfloat16* tmp = yin; yin = yout; yout = tmp;
    }

    // 5. final FC: out (8192 x 512) fp32
    gemm(yin, 1024, Wfc, 1024, fcb, nullptr, d_out, 512, 8192, 512, 1024, 0);
}
